// AlternatingDiffHead_1348619731545
// MI455X (gfx1250) — matrix operand order repaired and checked by reading, not compiled, not run on hardware
//
#include <hip/hip_runtime.h>
#include <hip/hip_bf16.h>

#define B_ 8
#define T_ 2048
#define C_ 1024
#define HS_ 128
#define BT_ (B_*T_)

typedef __bf16 bf16;
typedef __attribute__((ext_vector_type(16))) __bf16 v16bf;
typedef __attribute__((ext_vector_type(8)))  __bf16 v8bf;
typedef __attribute__((ext_vector_type(8)))  float  v8f;
typedef __attribute__((ext_vector_type(4)))  float  v4f;

__device__ __forceinline__ v8f wmma_bf16(v16bf a, v16bf b, v8f c) {
  // D = A(16x32 bf16) * B(32x16 bf16) + C(16x16 f32)
  return __builtin_amdgcn_wmma_f32_16x16x32_bf16(
      /*neg_a=*/false, a, /*neg_b=*/false, b,
      /*c_mod=*/(short)0, c, /*reuse_a=*/false, /*reuse_b=*/false);
}

__device__ __forceinline__ v16bf join8(v8bf lo, v8bf hi) {
  v16bf r;
#pragma unroll
  for (int i = 0; i < 8; ++i) { r[i] = lo[i]; r[8 + i] = hi[i]; }
  return r;
}

__device__ __forceinline__ v8f v8f_zero() {
  v8f r;
#pragma unroll
  for (int i = 0; i < 8; ++i) r[i] = 0.f;
  return r;
}

// ---------------------------------------------------------------------------
// Kernel 1: lambda scalars.  lam[0] = e0 + lam_init ; lam[1] = e1 - e0 + lam_init
// ---------------------------------------------------------------------------
__global__ void lam_kernel(const float* __restrict__ lq, const float* __restrict__ lk,
                           const int* __restrict__ lidx, float* __restrict__ lam) {
  __shared__ float red[256];
  const int t = threadIdx.x;      // t = n*128 + d
  const int d = t & 127;
  red[t] = __expf(lq[t] * lk[t]);
  __syncthreads();
  for (int s = 64; s > 0; s >>= 1) {
    if (d < s) red[t] += red[t + s];
    __syncthreads();
  }
  if (t == 0) {
    const float e0 = red[0]   * (1.0f / 128.0f);
    const float e1 = red[128] * (1.0f / 128.0f);
    const float li = (float)lidx[0];
    const float lam_init = 0.8f - 0.6f * __expf(-0.3f * (li - 1.0f));
    lam[0] = e0 + lam_init;
    lam[1] = e1 - e0 + lam_init;
  }
}

// ---------------------------------------------------------------------------
// Kernel 2: projections.
//   job 0/1 : q head 0/1  (RoPE, store bf16 row-major (n,b,t,d))
//   job 2/3 : k head 0/1  (RoPE, store bf16 row-major (n,b,t,d))
//   job 4   : v           (store bf16 transposed (b,d,t))
// Block = 256 threads (8 waves), computes 16 rows x ncols, K-loop of 32.
// ---------------------------------------------------------------------------
__global__ __launch_bounds__(256, 1)
void proj_kernel(const float* __restrict__ x, const float* __restrict__ Wq,
                 const float* __restrict__ Wk, const float* __restrict__ Wv,
                 bf16* __restrict__ qws, bf16* __restrict__ kws, bf16* __restrict__ vtws) {
  __shared__ bf16 ldsB[256 * 32];   // W chunk, column-major: [col][k]
  __shared__ bf16 ldsE[8 * 256];    // epilogue bounce, per-wave 16x16 tile

  const int job     = blockIdx.y;
  const int rowBase = blockIdx.x * 16;          // flattened (b,t) row
  const int tid     = threadIdx.x;
  const int wave    = tid >> 5;
  const int lane    = tid & 31;
  const int l15     = lane & 15;
  const int hi      = (lane >= 16) ? 1 : 0;

  const bool  isV    = (job == 4);
  const int   cshift = isV ? 8 : 7;             // ncols = 256 : 128 (power of 2)
  const int   cmask  = (1 << cshift) - 1;
  const float* W = isV ? Wv
                       : ((job < 2) ? (Wq + (size_t)job * C_ * HS_)
                                    : (Wk + (size_t)(job - 2) * C_ * HS_));

  v8f acc0 = v8f_zero();
  v8f acc1 = v8f_zero();            // second col tile (v job only)

  const float* xrow = x + (size_t)(rowBase + l15) * C_;
  const int    aoff = hi ? 16 : 0;  // A layout: lane>=16 holds K+16..K+31

  for (int kk = 0; kk < C_; kk += 32) {
    // stage W[kk..kk+31, :] transposed into LDS as bf16 (shift/mask indexing)
    for (int idx = tid; idx < (32 << cshift); idx += 256) {
      const int kr  = idx >> cshift;
      const int col = idx & cmask;
      ldsB[col * 32 + kr] = (bf16)W[(size_t)(kk + kr) << cshift | col];
    }
    __syncthreads();

    // A fragment from x (fp32 -> bf16): K = aoff+[0..15]
    const v4f f0 = *(const v4f*)(xrow + kk + aoff);
    const v4f f1 = *(const v4f*)(xrow + kk + aoff + 4);
    const v4f f2 = *(const v4f*)(xrow + kk + aoff + 8);
    const v4f f3 = *(const v4f*)(xrow + kk + aoff + 12);
    v16bf a;
#pragma unroll
    for (int i = 0; i < 4; ++i) {
      a[i]      = (bf16)f0[i];
      a[4 + i]  = (bf16)f1[i];
      a[8 + i]  = (bf16)f2[i];
      a[12 + i] = (bf16)f3[i];
    }

    {
      const int  col = (isV ? wave * 32 : wave * 16) + l15;
      const bf16* bp = &ldsB[col * 32 + (hi ? 16 : 0)];
      acc0 = wmma_bf16(a, join8(*(const v8bf*)bp, *(const v8bf*)(bp + 8)), acc0);
    }
    if (isV) {
      const int  col = wave * 32 + 16 + l15;
      const bf16* bp = &ldsB[col * 32 + (hi ? 16 : 0)];
      acc1 = wmma_bf16(a, join8(*(const v8bf*)bp, *(const v8bf*)(bp + 8)), acc1);
    }
    __syncthreads();   // protect ldsB for next K step
  }

  const int tseq = rowBase & (T_ - 1);   // position in sequence (T divides tiles)

  if (!isV) {
    // ---- RoPE epilogue + coalesced bf16 store via LDS bounce ----
    const int   n   = (job < 2) ? job : (job - 2);
    bf16* dst = ((job < 2) ? qws : kws) + ((size_t)n * BT_ + rowBase) * HS_;
    const int   colBase = wave * 16;
    const int   d   = colBase + l15;
    const int   i2  = d >> 1;
    const bool  odd = d & 1;
    // freq = 10000^(-2*i2/128) = 2^(-2*i2 * log2(1e4)/128)
    const float freq = exp2f(-(float)(2 * i2) * (13.287712379549449f / 128.0f));
#pragma unroll
    for (int r = 0; r < 8; ++r) {
      const int   lr  = r + (hi ? 8 : 0);
      const float ang = (float)(tseq + lr) * freq;
      float sn, cs;
      __sincosf(ang, &sn, &cs);   // hw v_sin/v_cos: ang <= 2048, no slow path needed
      const float self  = acc0[r];
      const float other = __shfl_xor(self, 1, 32);   // RoPE pair partner (d^1)
      const float val   = odd ? (other * sn + self * cs)
                              : (self * cs - other * sn);
      ldsE[wave * 256 + lr * 16 + l15] = (bf16)val;
    }
    __builtin_amdgcn_wave_barrier();   // in-order DS pipe orders write->read
    {
      const int lr2  = lane >> 1;
      const int half = lane & 1;
      const v8bf vv = *(const v8bf*)&ldsE[wave * 256 + lr2 * 16 + half * 8];
      *(v8bf*)(dst + (size_t)lr2 * HS_ + colBase + half * 8) = vv;
    }
  } else {
    // ---- v: store transposed (b, d, t); rows of accumulator are contiguous t ----
    const int b  = rowBase / T_;
#pragma unroll
    for (int sub = 0; sub < 2; ++sub) {
      const v8f acc = sub ? acc1 : acc0;
      const int d   = wave * 32 + sub * 16 + l15;
      v8bf pk;
#pragma unroll
      for (int r = 0; r < 8; ++r) pk[r] = (bf16)acc[r];
      *(v8bf*)(vtws + ((size_t)b * 256 + d) * T_ + tseq + (hi ? 8 : 0)) = pk;
    }
  }
}

// ---------------------------------------------------------------------------
// Kernel 3: flash attention, per-wave 16 q-rows, chunks of 32 keys.
// out = lam0 * softmax(q0 k0^T) @ v  -  lam1 * softmax(q1 k1^T) @ v
// ---------------------------------------------------------------------------
__global__ __launch_bounds__(128, 1)
void attn_kernel(const bf16* __restrict__ qws, const bf16* __restrict__ kws,
                 const bf16* __restrict__ vtws, const float* __restrict__ lam,
                 float* __restrict__ out) {
  __shared__ bf16 ldsP[4 * 16 * 32];    // per-wave P tile (C-layout -> A-layout)

  const int b    = blockIdx.y;
  const int wave = threadIdx.x >> 5;
  const int lane = threadIdx.x & 31;
  const int l15  = lane & 15;
  const int hi   = (lane >= 16) ? 1 : 0;
  const int t0   = blockIdx.x * 64 + wave * 16;
  const float scale = 0.08838834764831845f;   // 1/sqrt(128)
  const float lam0 = lam[0], lam1 = lam[1];
  bf16* myP = &ldsP[wave * 512];

  for (int n = 0; n < 2; ++n) {
    // Q fragments for rows t0..t0+15 (A layout), resident in registers
    const bf16* qrow = qws + ((size_t)(n * B_ + b) * T_ + t0 + l15) * HS_ + (hi ? 8 : 0);
    v16bf aq[4];
#pragma unroll
    for (int f = 0; f < 4; ++f)
      aq[f] = join8(*(const v8bf*)(qrow + f * 32),
                    *(const v8bf*)(qrow + f * 32 + 16));

    float m[8], l[8];
#pragma unroll
    for (int r = 0; r < 8; ++r) { m[r] = -1e30f; l[r] = 0.f; }
    v8f oh[16];
#pragma unroll
    for (int dt = 0; dt < 16; ++dt) oh[dt] = v8f_zero();

    const bf16* kbase = kws + (size_t)(n * B_ + b) * T_ * HS_;
    const bf16* vbase = vtws + (size_t)b * 256 * T_;

    const int nch = (t0 + 16 + 31) >> 5;      // causal: only s <= t0+15
    for (int c = 0; c < nch; ++c) {
      const int s0 = c * 32;

      // prefetch next chunk's K and V into cache (global_prefetch_b8)
      if (c + 1 < nch) {
        __builtin_prefetch(kbase + (size_t)(s0 + 32 + lane) * HS_, 0, 1);
        __builtin_prefetch(vbase + (size_t)(lane * 8) * T_ + s0 + 32, 0, 1);
      }

      // S = Q K^T : two 16x16 tiles, K-loop over HS (4 x 32)
      v8f st[2];
#pragma unroll
      for (int j = 0; j < 2; ++j) {
        v8f acc = v8f_zero();
        const bf16* krow = kbase + (size_t)(s0 + j * 16 + l15) * HS_ + (hi ? 8 : 0);
#pragma unroll
        for (int f = 0; f < 4; ++f) {
          const v16bf bk = join8(*(const v8bf*)(krow + f * 32),
                                 *(const v8bf*)(krow + f * 32 + 16));
          acc = wmma_bf16(aq[f], bk, acc);
        }
        st[j] = acc;
      }

      // online softmax: scale, causal mask, row max/sum via shfl within halves
      float mnew[8];
#pragma unroll
      for (int r = 0; r < 8; ++r) {
        const int tg = t0 + r + (hi ? 8 : 0);
        float x0 = st[0][r] * scale; if (s0 + l15 > tg)       x0 = -1e30f;
        float x1 = st[1][r] * scale; if (s0 + 16 + l15 > tg)  x1 = -1e30f;
        st[0][r] = x0; st[1][r] = x1;
        float mx = fmaxf(x0, x1);
#pragma unroll
        for (int msk = 1; msk < 16; msk <<= 1) mx = fmaxf(mx, __shfl_xor(mx, msk, 32));
        mnew[r] = fmaxf(m[r], mx);
      }
      float alpha[8];
#pragma unroll
      for (int r = 0; r < 8; ++r) {
        alpha[r] = __expf(m[r] - mnew[r]);
        const float p0 = __expf(st[0][r] - mnew[r]);
        const float p1 = __expf(st[1][r] - mnew[r]);
        float s = p0 + p1;
#pragma unroll
        for (int msk = 1; msk < 16; msk <<= 1) s += __shfl_xor(s, msk, 32);
        l[r] = l[r] * alpha[r] + s;
        m[r] = mnew[r];
        const int lr = r + (hi ? 8 : 0);
        myP[lr * 32 + l15]      = (bf16)p0;
        myP[lr * 32 + 16 + l15] = (bf16)p1;
      }
#pragma unroll
      for (int dt = 0; dt < 16; ++dt)
#pragma unroll
        for (int r = 0; r < 8; ++r) oh[dt][r] *= alpha[r];

      __builtin_amdgcn_wave_barrier();
      // re-read P as an A fragment (16x32 bf16)
      const bf16* pp = myP + l15 * 32 + (hi ? 8 : 0);
      const v16bf ap = join8(*(const v8bf*)pp, *(const v8bf*)(pp + 16));
      __builtin_amdgcn_wave_barrier();

      // O += P @ V : 16 output d-tiles, V streamed from transposed layout
      const bf16* vrow = vbase + (size_t)l15 * T_ + s0 + (hi ? 8 : 0);
#pragma unroll
      for (int dt = 0; dt < 16; ++dt) {
        const bf16* vp = vrow + (size_t)dt * 16 * T_;
        const v16bf bv = join8(*(const v8bf*)vp, *(const v8bf*)(vp + 16));
        oh[dt] = wmma_bf16(ap, bv, oh[dt]);
      }
    }

    // fold head result into out (head 0 writes, head 1 same-thread RMW)
    float linv[8];
#pragma unroll
    for (int r = 0; r < 8; ++r) linv[r] = __builtin_amdgcn_rcpf(l[r]);
    float* ob = out + ((size_t)b * T_ + t0 + (hi ? 8 : 0)) * 256 + l15;
    if (n == 0) {
#pragma unroll
      for (int dt = 0; dt < 16; ++dt)
#pragma unroll
        for (int r = 0; r < 8; ++r)
          ob[(size_t)r * 256 + dt * 16] = lam0 * linv[r] * oh[dt][r];
    } else {
#pragma unroll
      for (int dt = 0; dt < 16; ++dt)
#pragma unroll
        for (int r = 0; r < 8; ++r) {
          const float prev = ob[(size_t)r * 256 + dt * 16];
          ob[(size_t)r * 256 + dt * 16] = prev - lam1 * linv[r] * oh[dt][r];
        }
    }
  }
}

// ---------------------------------------------------------------------------
extern "C" void kernel_launch(void* const* d_in, const int* in_sizes, int n_in,
                              void* d_out, int out_size, void* d_ws, size_t ws_size,
                              hipStream_t stream) {
  (void)in_sizes; (void)n_in; (void)out_size; (void)ws_size;
  const float* x   = (const float*)d_in[0];
  const float* Wq  = (const float*)d_in[1];
  const float* Wk  = (const float*)d_in[2];
  const float* Wv  = (const float*)d_in[3];
  const float* lq  = (const float*)d_in[4];
  const float* lk  = (const float*)d_in[5];
  const int*   li  = (const int*)d_in[6];
  float* out = (float*)d_out;

  // workspace: [lam(2f) pad 256B][q bf16 2*BT*HS][k bf16 2*BT*HS][vT bf16 B*256*T]
  char*  w    = (char*)d_ws;
  float* lam  = (float*)w;
  bf16*  qws  = (bf16*)(w + 256);
  bf16*  kws  = qws + (size_t)2 * BT_ * HS_;
  bf16*  vtws = kws + (size_t)2 * BT_ * HS_;

  lam_kernel <<<1, 256, 0, stream>>>(lq, lk, li, lam);
  proj_kernel<<<dim3(BT_ / 16, 5), 256, 0, stream>>>(x, Wq, Wk, Wv, qws, kws, vtws);
  attn_kernel<<<dim3(T_ / 64, B_), 128, 0, stream>>>(qws, kws, vtws, lam, out);
}
